// VGG_GCN_graph_44487271252380
// MI455X (gfx1250) — matrix operand order repaired and checked by reading, not compiled, not run on hardware
//
#include <hip/hip_runtime.h>
#include <hip/hip_bf16.h>
#include <hip/hip_fp16.h>

typedef __attribute__((ext_vector_type(16))) _Float16 v16h;
typedef __attribute__((ext_vector_type(8)))  float    v8f;

#define AST 40   // LDS row stride (halves): 80B = 16B multiple -> b128-friendly
#define BST 40

// ---------------------------------------------------------------- convert (4 elems/thread)
__global__ void k_f32_to_f16(const float* __restrict__ s, _Float16* __restrict__ d, int n) {
  int i = (blockIdx.x * blockDim.x + threadIdx.x) * 4;
  if (i >= n) return;
  float4 f = *(const float4*)(s + i);
  union { _Float16 h[4]; uint2 u; } o;
  o.h[0] = (_Float16)f.x; o.h[1] = (_Float16)f.y;
  o.h[2] = (_Float16)f.z; o.h[3] = (_Float16)f.w;
  *(uint2*)(d + i) = o.u;
}

// ---------------------------------------------------------------- 2x2 maxpool (f16, NCHW, pow2 dims)
__global__ void k_maxpool2(const _Float16* __restrict__ x, _Float16* __restrict__ y,
                           int lgC, int lgH, int lgW, int n) {
  int i = blockIdx.x * blockDim.x + threadIdx.x;
  if (i >= n) return;
  const int lgWO = lgW - 1, lgHO = lgH - 1;
  int wo = i & ((1 << lgWO) - 1);
  int t  = i >> lgWO;
  int ho = t & ((1 << lgHO) - 1); t >>= lgHO;
  int c  = t & ((1 << lgC) - 1);
  int b  = t >> lgC;
  const int W = 1 << lgW;
  const _Float16* p = x + ((((size_t)((b << lgC) + c) << lgH) + (ho << 1)) << lgW) + (wo << 1);
  float a0 = (float)p[0], a1 = (float)p[1];
  float a2 = (float)p[W], a3 = (float)p[W + 1];
  y[i] = (_Float16)fmaxf(fmaxf(a0, a1), fmaxf(a2, a3));
}

// ---------------------------------------------------------------- graph_gt (n == 512)
__global__ void k_graph_gt(const int* __restrict__ t, float* __restrict__ g) {
  int i = blockIdx.x * blockDim.x + threadIdx.x;
  if (i >= 512 * 512) return;
  int r = i >> 9, c = i & 511;
  g[i] = (t[r] == t[c]) ? 1.0f : 0.0f;
}

// ---------------------------------------------------------------- fragment gathers (2x ds_load_b128 each)
// A frag (16x32 f16): lane m=l&15; lanes 0-15 hold K 0-7 & 16-23, lanes 16-31 hold K 8-15 & 24-31.
// B frag (32x16 f16): lane n=l&15; lanes 0-15 hold K 0-7 & 16-23, lanes 16-31 hold K 8-15 & 24-31.
union HFrag { v16h v; uint4 q[2]; };

__device__ __forceinline__ v16h load_frag_a(const _Float16* As, int rowBase, int lane) {
  HFrag f;
  const int m  = lane & 15;
  const int kb = (lane >> 4) << 3;
  const _Float16* Ar = &As[(rowBase + m) * AST];
  f.q[0] = *(const uint4*)&Ar[kb];
  f.q[1] = *(const uint4*)&Ar[16 + kb];
  return f.v;
}

__device__ __forceinline__ v16h load_frag_b(const _Float16* Bs, int colBase, int lane) {
  HFrag f;
  const int n   = lane & 15;
  const int kb2 = (lane >> 4) << 3;
  const _Float16* Br = &Bs[(colBase + n) * BST];
  f.q[0] = *(const uint4*)&Br[kb2];
  f.q[1] = *(const uint4*)&Br[16 + kb2];
  return f.v;
}

#define WMMA16(A, B, C) __builtin_amdgcn_wmma_f32_16x16x32_f16(false, A, false, B, (short)0, C, false, false)

// ---------------------------------------------------------------- implicit-GEMM conv3x3 pad1 + bias + relu
// 64(O) x 128(pixels) macro-tile, 8 waves 2x4, 32x32/wave = 4 WMMA per K-step.
// Double-buffered LDS: global loads of K-slab k0+32 overlap the WMMAs of slab k0.
__global__ __launch_bounds__(256)
void k_conv_wmma(const _Float16* __restrict__ X, const _Float16* __restrict__ Wt,
                 const float* __restrict__ bias, _Float16* __restrict__ Y,
                 int C, int O, int lgHW, int lgW, int Ktot) {
  __shared__ __align__(16) _Float16 As[2 * 64 * AST];
  __shared__ __align__(16) _Float16 Bs[2 * 128 * BST];
  const int tid  = threadIdx.x;
  const int lane = tid & 31;
  const int wv   = tid >> 5;
  const int mi2  = wv >> 2;            // 0..1 : 32-row block
  const int ni2  = wv & 3;             // 0..3 : 32-col block
  const int p0 = blockIdx.x * 128;
  const int o0 = blockIdx.y * 64;
  const int HW = 1 << lgHW;
  const int W  = 1 << lgW;
  const int H  = HW >> lgW;
  const bool kvec = (Ktot & 31) == 0;

  const int am  = tid >> 2;            // A row 0..63
  const int akg = (tid & 3) << 3;      // A k-chunk (8 halves)
  const int bk  = tid & 31;            // B k (fixed per thread)
  const int bp  = tid >> 5;            // B base pixel (stride 8)

  v8f acc[2][2] = {};
  uint4 ra;                            // A slab (vector path)
  _Float16 ras[8];                     // A slab (scalar path, layer 0)
  _Float16 rb[16];                     // B slab (im2col gather)

  auto loadregs = [&](int k0) {
    if (kvec) {
      ra = *(const uint4*)&Wt[(size_t)(o0 + am) * Ktot + k0 + akg];
      if (k0 + 32 < Ktot)
        __builtin_prefetch(&Wt[(size_t)(o0 + am) * Ktot + k0 + 32 + akg], 0, 1);
    } else {
#pragma unroll
      for (int e = 0; e < 8; ++e) {
        int q = tid + 256 * e;
        int m = q >> 5, k = q & 31;
        int kk = k0 + k;
        ras[e] = (kk < Ktot) ? Wt[(size_t)(o0 + m) * Ktot + kk] : (_Float16)0.f;
      }
    }
    const int kk = k0 + bk;
    const int c  = kk / 9;
    const int r9 = kk - c * 9;
    const int kh = r9 / 3;
    const int kw = r9 - kh * 3;
    const bool kok = kk < Ktot;
    const _Float16* Xc = X + ((size_t)c << lgHW);
#pragma unroll
    for (int e = 0; e < 16; ++e) {
      int pg = p0 + bp + 8 * e;
      int b  = pg >> lgHW;
      int hw = pg & (HW - 1);
      int h  = hw >> lgW;
      int w  = hw & (W - 1);
      int ih = h + kh - 1, iw = w + kw - 1;
      _Float16 v = (_Float16)0.f;
      if (kok && (unsigned)ih < (unsigned)H && (unsigned)iw < (unsigned)W)
        v = Xc[(((size_t)b * C) << lgHW) + (ih << lgW) + iw];
      rb[e] = v;
    }
  };
  auto commit = [&](int buf) {
    _Float16* Asb = As + buf * (64 * AST);
    _Float16* Bsb = Bs + buf * (128 * BST);
    if (kvec) {
      *(uint4*)&Asb[am * AST + akg] = ra;
    } else {
#pragma unroll
      for (int e = 0; e < 8; ++e) {
        int q = tid + 256 * e;
        int m = q >> 5, k = q & 31;
        Asb[m * AST + k] = ras[e];
      }
    }
#pragma unroll
    for (int e = 0; e < 16; ++e)
      Bsb[(bp + 8 * e) * BST + bk] = rb[e];
  };
  auto compute = [&](int buf) {
    const _Float16* Asb = As + buf * (64 * AST);
    const _Float16* Bsb = Bs + buf * (128 * BST);
    v16h a0 = load_frag_a(Asb, mi2 * 32 + 0,  lane);
    v16h a1 = load_frag_a(Asb, mi2 * 32 + 16, lane);
    v16h b0 = load_frag_b(Bsb, ni2 * 32 + 0,  lane);
    v16h b1 = load_frag_b(Bsb, ni2 * 32 + 16, lane);
    acc[0][0] = WMMA16(a0, b0, acc[0][0]);
    acc[0][1] = WMMA16(a0, b1, acc[0][1]);
    acc[1][0] = WMMA16(a1, b0, acc[1][0]);
    acc[1][1] = WMMA16(a1, b1, acc[1][1]);
  };

  // software pipeline: one barrier per K-step, global loads overlap WMMA
  loadregs(0);
  commit(0);
  __syncthreads();
  int buf = 0;
  for (int k0 = 32; k0 < Ktot; k0 += 32) {
    loadregs(k0);
    compute(buf);
    commit(buf ^ 1);
    __syncthreads();
    buf ^= 1;
  }
  compute(buf);

  // ---- epilogue: bias + relu, scatter f16 NCHW
  const int mrow = (lane >> 4) * 8;
#pragma unroll
  for (int sm = 0; sm < 2; ++sm) {
    float bb[8];
    *(float4*)&bb[0] = *(const float4*)&bias[o0 + mi2 * 32 + sm * 16 + mrow];
    *(float4*)&bb[4] = *(const float4*)&bias[o0 + mi2 * 32 + sm * 16 + mrow + 4];
#pragma unroll
    for (int sn = 0; sn < 2; ++sn) {
      int p  = p0 + ni2 * 32 + sn * 16 + (lane & 15);
      int b  = p >> lgHW;
      int hw = p & (HW - 1);
      _Float16* Yb = Y + (((size_t)b * O) << lgHW) + hw;
#pragma unroll
      for (int r = 0; r < 8; ++r) {
        int o = o0 + mi2 * 32 + sm * 16 + mrow + r;
        Yb[(size_t)o << lgHW] = (_Float16)fmaxf(acc[sm][sn][r] + bb[r], 0.f);
      }
    }
  }
}

// ---------------------------------------------------------------- dense GEMM, 64x128 tile, double-buffered
// EPI: 0 = bias+relu -> f16 ; 1 = bias -> f16 ; 2 = sigmoid -> f32 ; 3 = bias -> f32 (N-bounded)
// TRANSB: B row-major [N][K] (low @ low^T)
template <int EPI, bool TRANSB>
__global__ __launch_bounds__(256)
void k_gemm_wmma(const _Float16* __restrict__ A, const _Float16* __restrict__ B,
                 const float* __restrict__ bias, float* __restrict__ outF,
                 _Float16* __restrict__ outH, int M, int N, int K) {
  __shared__ __align__(16) _Float16 As[2 * 64 * AST];
  __shared__ __align__(16) _Float16 Bs[2 * 128 * BST];   // [n][k]
  const int tid  = threadIdx.x;
  const int lane = tid & 31;
  const int wv   = tid >> 5;
  const int mi2  = wv >> 2;
  const int ni2  = wv & 3;
  const int n0 = blockIdx.x * 128;
  const int m0 = blockIdx.y * 64;

  const int am  = tid >> 2;
  const int akg = (tid & 3) << 3;

  v8f acc[2][2] = {};
  uint4 ra;
  uint4 rbv[2];
  _Float16 rbs[16];

  auto loadregs = [&](int k0) {
    ra = *(const uint4*)&A[(size_t)(m0 + am) * K + k0 + akg];
    if (TRANSB) {
#pragma unroll
      for (int e = 0; e < 2; ++e) {
        int q = tid + 256 * e;
        int n = q >> 2, kg = (q & 3) << 3;
        rbv[e] = *(const uint4*)&B[(size_t)(n0 + n) * K + k0 + kg];
      }
    } else {
#pragma unroll
      for (int e = 0; e < 16; ++e) {
        int q = tid + 256 * e;
        int n = q >> 5, k = q & 31;
        int gn = n0 + n;
        rbs[e] = (gn < N) ? B[(size_t)(k0 + k) * N + gn] : (_Float16)0.f;
      }
    }
  };
  auto commit = [&](int buf) {
    _Float16* Asb = As + buf * (64 * AST);
    _Float16* Bsb = Bs + buf * (128 * BST);
    *(uint4*)&Asb[am * AST + akg] = ra;
    if (TRANSB) {
#pragma unroll
      for (int e = 0; e < 2; ++e) {
        int q = tid + 256 * e;
        int n = q >> 2, kg = (q & 3) << 3;
        *(uint4*)&Bsb[n * BST + kg] = rbv[e];
      }
    } else {
#pragma unroll
      for (int e = 0; e < 16; ++e) {
        int q = tid + 256 * e;
        int n = q >> 5, k = q & 31;
        Bsb[n * BST + k] = rbs[e];
      }
    }
  };
  auto compute = [&](int buf) {
    const _Float16* Asb = As + buf * (64 * AST);
    const _Float16* Bsb = Bs + buf * (128 * BST);
    v16h a0 = load_frag_a(Asb, mi2 * 32 + 0,  lane);
    v16h a1 = load_frag_a(Asb, mi2 * 32 + 16, lane);
    v16h b0 = load_frag_b(Bsb, ni2 * 32 + 0,  lane);
    v16h b1 = load_frag_b(Bsb, ni2 * 32 + 16, lane);
    acc[0][0] = WMMA16(a0, b0, acc[0][0]);
    acc[0][1] = WMMA16(a0, b1, acc[0][1]);
    acc[1][0] = WMMA16(a1, b0, acc[1][0]);
    acc[1][1] = WMMA16(a1, b1, acc[1][1]);
  };

  loadregs(0);
  commit(0);
  __syncthreads();
  int buf = 0;
  for (int k0 = 32; k0 < K; k0 += 32) {
    loadregs(k0);
    compute(buf);
    commit(buf ^ 1);
    __syncthreads();
    buf ^= 1;
  }
  compute(buf);

  const int mrow = (lane >> 4) * 8;
#pragma unroll
  for (int sn = 0; sn < 2; ++sn) {
    int gn = n0 + ni2 * 32 + sn * 16 + (lane & 15);
#pragma unroll
    for (int sm = 0; sm < 2; ++sm) {
#pragma unroll
      for (int r = 0; r < 8; ++r) {
        int gm = m0 + mi2 * 32 + sm * 16 + mrow + r;
        float v = acc[sm][sn][r];
        if (EPI == 0) {
          v = fmaxf(v + bias[gn], 0.f);
          outH[(size_t)gm * N + gn] = (_Float16)v;
        } else if (EPI == 1) {
          outH[(size_t)gm * N + gn] = (_Float16)(v + bias[gn]);
        } else if (EPI == 2) {
          outF[(size_t)gm * N + gn] = 1.f / (1.f + __expf(-v));
        } else {
          if (gn < N) outF[(size_t)gm * N + gn] = v + bias[gn];
        }
      }
    }
  }
}

// ---------------------------------------------------------------- host side
struct ConvDesc { int C, O, H, W, pool; };
static const ConvDesc CD[13] = {
  {  3,  64, 32, 32, 0}, { 64,  64, 32, 32, 1},
  { 64, 128, 16, 16, 0}, {128, 128, 16, 16, 1},
  {128, 256,  8,  8, 0}, {256, 256,  8,  8, 0}, {256, 256,  8,  8, 1},
  {256, 512,  4,  4, 0}, {512, 512,  4,  4, 0}, {512, 512,  4,  4, 1},
  {512, 512,  2,  2, 0}, {512, 512,  2,  2, 0}, {512, 512,  2,  2, 1},
};

extern "C" void kernel_launch(void* const* d_in, const int* in_sizes, int n_in,
                              void* d_out, int out_size, void* d_ws, size_t ws_size,
                              hipStream_t stream) {
  (void)in_sizes; (void)n_in; (void)out_size; (void)ws_size;
  const int BATCH = 512;

  char* ws = (char*)d_ws;
  size_t off = 0;
  auto carve = [&](size_t bytes) -> char* {
    char* p = ws + off;
    off += (bytes + 255) & ~(size_t)255;
    return p;
  };
  const size_t ACT_ELEMS = (size_t)BATCH * 64 * 32 * 32;
  _Float16* actA = (_Float16*)carve(ACT_ELEMS * 2);
  _Float16* actB = (_Float16*)carve(ACT_ELEMS * 2);
  _Float16* wh[13];
  for (int i = 0; i < 13; ++i)
    wh[i] = (_Float16*)carve((size_t)CD[i].O * CD[i].C * 9 * 2);
  _Float16* l1h  = (_Float16*)carve((size_t)512 * 256 * 2);
  _Float16* l2h  = (_Float16*)carve((size_t)256 * 128 * 2);
  _Float16* clh  = (_Float16*)carve((size_t)512 * 10 * 2);
  _Float16* hH   = (_Float16*)carve((size_t)BATCH * 256 * 2);
  _Float16* lowH = (_Float16*)carve((size_t)BATCH * 128 * 2);

  auto cvt = [&](const void* s, _Float16* d, int n) {
    int thr = (n + 3) / 4;
    k_f32_to_f16<<<(thr + 255) / 256, 256, 0, stream>>>((const float*)s, d, n);
  };

  cvt(d_in[0], actA, BATCH * 3 * 32 * 32);
  for (int i = 0; i < 13; ++i)
    cvt(d_in[2 + 2 * i], wh[i], CD[i].O * CD[i].C * 9);
  cvt(d_in[28], l1h, 512 * 256);
  cvt(d_in[30], l2h, 256 * 128);
  cvt(d_in[32], clh, 512 * 10);

  _Float16* cur = actA;
  _Float16* nxt = actB;
  for (int i = 0; i < 13; ++i) {
    const ConvDesc& L = CD[i];
    int lgW  = __builtin_ctz(L.W);
    int lgHW = lgW + __builtin_ctz(L.H);
    int HW = L.H * L.W;
    int Ktot = L.C * 9;
    dim3 grid((BATCH * HW) / 128, L.O / 64);
    k_conv_wmma<<<grid, 256, 0, stream>>>(cur, wh[i], (const float*)d_in[3 + 2 * i],
                                          nxt, L.C, L.O, lgHW, lgW, Ktot);
    { _Float16* t = cur; cur = nxt; nxt = t; }
    if (L.pool) {
      int n = BATCH * L.O * (L.H / 2) * (L.W / 2);
      k_maxpool2<<<(n + 255) / 256, 256, 0, stream>>>(cur, nxt, __builtin_ctz(L.O),
                                                      __builtin_ctz(L.H), lgW, n);
      { _Float16* t = cur; cur = nxt; nxt = t; }
    }
  }
  // cur == feat, f16 [512][512]

  float* outP = (float*)d_out;            // [512,10]
  float* gpre = outP + 512 * 10;          // [512,512]
  float* ggt  = gpre + 512 * 512;         // [512,512]

  k_gemm_wmma<0, false><<<dim3(2, 8), 256, 0, stream>>>(
      cur, l1h, (const float*)d_in[29], nullptr, hH, 512, 256, 512);
  k_gemm_wmma<1, false><<<dim3(1, 8), 256, 0, stream>>>(
      hH, l2h, (const float*)d_in[31], nullptr, lowH, 512, 128, 256);
  k_gemm_wmma<2, true><<<dim3(4, 8), 256, 0, stream>>>(
      lowH, lowH, nullptr, gpre, nullptr, 512, 512, 128);
  k_gemm_wmma<3, false><<<dim3(1, 8), 256, 0, stream>>>(
      cur, clh, (const float*)d_in[33], outP, nullptr, 512, 10, 512);
  k_graph_gt<<<(512 * 512 + 255) / 256, 256, 0, stream>>>((const int*)d_in[1], ggt);
}
